// AUGRU_modified_49984829391008
// MI455X (gfx1250) — matrix operand order repaired and checked by reading, not compiled, not run on hardware
//
#include <hip/hip_runtime.h>
#include <hip/hip_bf16.h>

// ---------------------------------------------------------------------------
// AUGRU fused recurrent kernel for gfx1250 (MI455X).
//   T=200, B=1024, D=128 (fixed by reference shapes).
// Strategy (v2):
//   * g[t,b] = att[t,b] * mask[b,t] premerged (setup kernel, into d_ws)
//   * 64 workgroups x 8 waves; each WG owns 16 batch rows for the whole T loop.
//   * fp32 master state lives in registers (C-fragment layout, 8 VGPRs/lane).
//   * ALL weight B-fragments live in REGISTERS for the whole T loop:
//     24 fragments x v16bf = 192 VGPRs per wave (gfx1250 allows up to 1024).
//     This removes ~75% of the per-step LDS traffic that bounded v1.
//   * LDS holds only the 16x256 bf16 A buffer ([x_t | s], 8 KB) used to
//     exchange the new state between waves each step.
//   * Per step: stage x_t (bf16) -> LDS, barrier, 16 ds_load_b128 for A
//     fragments, 24 v_wmma_f32_16x16x32_bf16, barrier,
//     s += g * sigmoid(u) * (tanh(xc + r*sc) - s), write bf16 s back.
// ---------------------------------------------------------------------------

typedef __bf16 v16bf __attribute__((ext_vector_type(16)));
typedef __bf16 v8bf  __attribute__((ext_vector_type(8)));
typedef float  v8f   __attribute__((ext_vector_type(8)));
typedef float  v4f   __attribute__((ext_vector_type(4)));

#define TT 200
#define BB 1024
#define DD 128

__device__ __forceinline__ v16bf cat8(v8bf lo, v8bf hi) {
  return __builtin_shufflevector(lo, hi, 0,1,2,3,4,5,6,7,8,9,10,11,12,13,14,15);
}
__device__ __forceinline__ float fast_sigmoid(float x) {
  return 1.0f / (1.0f + __expf(-x));
}
__device__ __forceinline__ float fast_tanh(float x) {
  return 1.0f - 2.0f / (__expf(2.0f * x) + 1.0f);
}

// g[t*B + b] = att[t*B + b] * mask[b*T + t]
__global__ void augru_fuse_gate(const float* __restrict__ att,
                                const float* __restrict__ mask,
                                float* __restrict__ g) {
  int idx = blockIdx.x * blockDim.x + threadIdx.x;
  if (idx < TT * BB) {
    int t = idx / BB;
    int b = idx - t * BB;
    g[idx] = att[idx] * mask[b * TT + t];
  }
}

__global__ void __launch_bounds__(256, 1) augru_recurrent(
    const float* __restrict__ xin,    // [T,B,D]
    const float* __restrict__ state,  // [B,D]
    const float* __restrict__ g,      // [T,B] premerged att*mask
    const float* __restrict__ Wau, const float* __restrict__ bau,
    const float* __restrict__ Wbu,
    const float* __restrict__ War, const float* __restrict__ bar,
    const float* __restrict__ Wbr,
    const float* __restrict__ Wac, const float* __restrict__ bac,
    const float* __restrict__ Wbc,
    float* __restrict__ out)          // [B,D]
{
  // A buffer: 16 rows x 256 cols bf16 ([x_t | s]) = 8 KB
  __shared__ __align__(32) __bf16 aBuf[16 * 256];

  const int tid  = threadIdx.x;
  const int wave = tid >> 5;
  const int lane = tid & 31;
  const int h    = lane >> 4;   // lane group (0: lanes 0-15, 1: lanes 16-31)
  const int ln   = lane & 15;
  const int rowbase = blockIdx.x * 16;

  // --------------------------------------------------------------------
  // One-time: load this wave's 24 weight B-fragments straight from global
  // into REGISTERS in WMMA B-fragment layout, where they stay for all T.
  // Logical weight matrix Wcat is 256(K) x 384(N):
  //   rows 0..127 = x-projection, rows 128..255 = s-projection
  //   frag q = 0..7  : u block (Wau over Wbu),  k-tile q&7
  //   frag q = 8..15 : r block (War over Wbr)
  //   frag q = 16..23: c block: k-tiles 0..3 = Wac (x), 4..7 = Wbc (s);
  //                    the zero cross-blocks are never touched.
  // B fragment (32x16 bf16): lane ln = N, element p (0..15) holds
  //   K = kt*32 + 16*h + p.   Column within D is j = wave*16 + ln.
  // --------------------------------------------------------------------
  const int colj = wave * 16 + ln;      // output feature column (N)
  v16bf breg[24];
  #pragma unroll
  for (int q = 0; q < 24; ++q) {
    const int blk = q >> 3;             // 0=u 1=r 2=c
    const int kt  = q & 7;
    const float* Wa = (blk == 0) ? Wau : (blk == 1) ? War : Wac;
    const float* Wb = (blk == 0) ? Wbu : (blk == 1) ? Wbr : Wbc;
    v16bf tmp;
    #pragma unroll
    for (int p = 0; p < 16; ++p) {
      int k = kt * 32 + h * 16 + p;
      float v = (k < 128) ? Wa[k * DD + colj] : Wb[(k - 128) * DD + colj];
      tmp[p] = (__bf16)v;
    }
    breg[q] = tmp;
  }

  // --------------------------------------------------------------------
  // Persistent fp32 state in C-fragment layout:
  //   acc VGPR i: lanes 0-15 -> row i, lanes 16-31 -> row 8+i; col = colj.
  // --------------------------------------------------------------------
  float s[8];
  #pragma unroll
  for (int i = 0; i < 8; ++i) {
    int row = rowbase + h * 8 + i;
    s[i] = state[row * DD + colj];
    aBuf[(h * 8 + i) * 256 + 128 + colj] = (__bf16)s[i];   // stage s0 (bf16)
  }
  const float bub = bau[colj];
  const float brb = bar[colj];
  const float bcb = bac[colj];

  for (int t = 0; t < TT; ++t) {
    // ---- stage x_t -> aBuf[:, 0:128] as bf16 (row-major A layout) ----
    {
      const int row = tid >> 4;              // 0..15
      const int c8  = (tid & 15) << 3;       // 0,8,...,120
      const float* src = xin + ((size_t)t * BB + rowbase + row) * DD + c8;
      v4f x0 = *(const v4f*)src;
      v4f x1 = *(const v4f*)(src + 4);
      __bf16* dst = &aBuf[row * 256 + c8];
      #pragma unroll
      for (int e = 0; e < 4; ++e) {
        dst[e]     = (__bf16)x0[e];
        dst[4 + e] = (__bf16)x1[e];
      }
      if (t + 1 < TT)
        __builtin_prefetch(src + (size_t)BB * DD, 0, 1);   // global_prefetch_b8
    }
    __syncthreads();

    // ---- load A fragments (16-bit A layout: lane ln = M; elems 0..15 =
    //      K kt*32+16h.., contiguous, matching B) ----
    v16bf a[8];
    #pragma unroll
    for (int kt = 0; kt < 8; ++kt) {
      const __bf16* ar = &aBuf[ln * 256 + kt * 32 + 16 * h];
      a[kt] = cat8(*(const v8bf*)ar, *(const v8bf*)(ar + 8));
    }

    // ---- per-row merged gate scalar g = att * mask (broadcast loads) ----
    const float* gp = g + (size_t)t * BB + rowbase + h * 8;
    v4f g0 = *(const v4f*)gp;
    v4f g1 = *(const v4f*)(gp + 4);

    // ---- GEMM: wave computes columns colj of u, r, xc, sc.
    //      Biases are folded into the accumulator init (broadcast over
    //      the 8 C VGPRs is exact: same column for all of them). ----
    v8f accU, accR, accXC, accSC;
    #pragma unroll
    for (int i = 0; i < 8; ++i) {
      accU[i] = bub; accR[i] = brb; accXC[i] = bcb; accSC[i] = 0.0f;
    }
    #pragma unroll
    for (int kt = 0; kt < 8; ++kt) {
      accU = __builtin_amdgcn_wmma_f32_16x16x32_bf16(false, a[kt], false, breg[kt],
                                                     (short)0, accU, false, false);
      accR = __builtin_amdgcn_wmma_f32_16x16x32_bf16(false, a[kt], false, breg[8 + kt],
                                                     (short)0, accR, false, false);
    }
    #pragma unroll
    for (int kt = 0; kt < 4; ++kt) {          // x part -> Wac
      accXC = __builtin_amdgcn_wmma_f32_16x16x32_bf16(false, a[kt], false, breg[16 + kt],
                                                      (short)0, accXC, false, false);
    }
    #pragma unroll
    for (int kt = 4; kt < 8; ++kt) {          // s part -> Wbc
      accSC = __builtin_amdgcn_wmma_f32_16x16x32_bf16(false, a[kt], false, breg[16 + kt],
                                                      (short)0, accSC, false, false);
    }

    __syncthreads();   // all aBuf reads done; safe to overwrite x / s regions

    // ---- gates + state update:  s += g * u * (c - s)  ----
    #pragma unroll
    for (int i = 0; i < 8; ++i) {
      float gg = (i < 4) ? g0[i] : g1[i - 4];
      float u  = fast_sigmoid(accU[i]);
      float r  = fast_sigmoid(accR[i]);
      float c  = fast_tanh(accXC[i] + r * accSC[i]);
      s[i] += gg * u * (c - s[i]);
      aBuf[(h * 8 + i) * 256 + 128 + colj] = (__bf16)s[i];  // bf16 s for next A
    }
  }

  // ---- write final state ----
  #pragma unroll
  for (int i = 0; i < 8; ++i)
    out[(size_t)(rowbase + h * 8 + i) * DD + colj] = s[i];
}

extern "C" void kernel_launch(void* const* d_in, const int* in_sizes, int n_in,
                              void* d_out, int out_size, void* d_ws, size_t ws_size,
                              hipStream_t stream) {
  (void)in_sizes; (void)n_in; (void)out_size; (void)ws_size;
  const float* xin   = (const float*)d_in[0];
  const float* state = (const float*)d_in[1];
  const float* att   = (const float*)d_in[2];   // [T,B,1]
  const float* mask  = (const float*)d_in[3];   // [B,T]
  // d_in[4] = max_len scalar (shapes are compile-time constants here)
  const float* Wau = (const float*)d_in[5];
  const float* bau = (const float*)d_in[6];
  const float* Wbu = (const float*)d_in[7];
  const float* War = (const float*)d_in[8];
  const float* bar = (const float*)d_in[9];
  const float* Wbr = (const float*)d_in[10];
  const float* Wac = (const float*)d_in[11];
  const float* bac = (const float*)d_in[12];
  const float* Wbc = (const float*)d_in[13];

  float* g = (float*)d_ws;   // TT*BB floats = 819200 bytes of scratch

  augru_fuse_gate<<<(TT * BB + 255) / 256, 256, 0, stream>>>(att, mask, g);

  augru_recurrent<<<BB / 16, 256, 0, stream>>>(
      xin, state, g, Wau, bau, Wbu, War, bar, Wbr, Wac, bac, Wbc,
      (float*)d_out);
}
